// SelfAttention_50629074486160
// MI455X (gfx1250) — compile-verified, matrix operand order checked by reading, not run on hardware
//
#include <hip/hip_runtime.h>
#include <hip/hip_bf16.h>
#include <math.h>

typedef __attribute__((ext_vector_type(16))) __bf16 v16bf;
typedef __attribute__((ext_vector_type(8)))  __bf16 v8bf;
typedef __attribute__((ext_vector_type(4)))  __bf16 v4bf;
typedef __attribute__((ext_vector_type(8)))  float  v8f;

constexpr int B_ = 64, S_ = 2048, D_ = 768, A_ = 256;
constexpr int ROWS   = B_ * S_;     // 131072 rows of the (B*S, D) matrix
constexpr int KT     = 32;          // K tile (bf16 WMMA depth)
constexpr int NKT    = D_ / KT;     // 24 k-steps
constexpr int NCT    = A_ / 16;     // 16 column tiles (A = 256)
constexpr int WAVES  = 8;           // waves per workgroup (wave32)
constexpr int WGROWS = WAVES * 16;  // 128 rows per workgroup

// ---------------------------------------------------------------------------
// Kernel 1: ait_raw[b,s] = exp( sum_a tanh( (x W)[row,a] + b[a] ) * u[a] ) * mask
// One wave = one 16-row strip; 16 WMMA column tiles held in 128 acc VGPRs.
// x read from HBM exactly once (f32 -> bf16 in-register); W stays L2-resident.
// ---------------------------------------------------------------------------
__global__ __launch_bounds__(256) void attn_scores_kernel(
    const float* __restrict__ x, const float* __restrict__ W,
    const float* __restrict__ bias, const float* __restrict__ u,
    const int* __restrict__ mask, float* __restrict__ ait)
{
    __shared__ __align__(16) __bf16 xs[WGROWS][KT];  //  8 KB: x k-slab (bf16)
    __shared__ __align__(16) __bf16 ws[A_][KT];      // 16 KB: W k-slab, transposed [col][k]

    const int tid  = threadIdx.x;
    const int lane = tid & 31;
    const int wave = tid >> 5;
    const long rowbase = (long)blockIdx.x * WGROWS;

    v8f acc[NCT];
#pragma unroll
    for (int c = 0; c < NCT; ++c) acc[c] = v8f{0.f,0.f,0.f,0.f,0.f,0.f,0.f,0.f};

    const int m  = lane & 15;          // row-in-strip / col-in-tile
    const int kb = (lane >> 4) * 8;    // K base per lane half (0 or 8)

    for (int kt = 0; kt < NKT; ++kt) {
        const int k0 = kt * KT;
        __syncthreads();

        // Stage x slab: 128 rows x 32 k. Each thread: 2 x (two b128 global
        // loads -> 8 bf16 cvt -> one ds_store_b128).
#pragma unroll
        for (int i = 0; i < (WGROWS * (KT / 8)) / 256; ++i) {   // 2 iters
            int e  = i * 256 + tid;
            int r  = e >> 2;            // row 0..127
            int ko = (e & 3) * 8;       // k-octet base 0,8,16,24
            const float* src = &x[(rowbase + r) * D_ + (k0 + ko)];
            float4 f0 = *(const float4*)(src);
            float4 f1 = *(const float4*)(src + 4);
            v8bf p;
            p[0]=(__bf16)f0.x; p[1]=(__bf16)f0.y; p[2]=(__bf16)f0.z; p[3]=(__bf16)f0.w;
            p[4]=(__bf16)f1.x; p[5]=(__bf16)f1.y; p[6]=(__bf16)f1.z; p[7]=(__bf16)f1.w;
            *(v8bf*)&xs[r][ko] = p;
        }

        // Stage W slab transposed: ws[col][kk] = W[k0+kk][col]. Each thread:
        // 4 coalesced f32 loads along col -> packed ds_store_b64.
#pragma unroll
        for (int i = 0; i < (A_ * (KT / 4)) / 256; ++i) {       // 8 iters
            int e   = i * 256 + tid;
            int col = e & 255;
            int kq  = (e >> 8) * 4;     // k-quad base 0,4,...,28
            float w0 = W[(long)(k0 + kq + 0) * A_ + col];
            float w1 = W[(long)(k0 + kq + 1) * A_ + col];
            float w2 = W[(long)(k0 + kq + 2) * A_ + col];
            float w3 = W[(long)(k0 + kq + 3) * A_ + col];
            v4bf p;
            p[0]=(__bf16)w0; p[1]=(__bf16)w1; p[2]=(__bf16)w2; p[3]=(__bf16)w3;
            *(v4bf*)&ws[col][kq] = p;
        }
        __syncthreads();

        // A fragment: row = wave*16 + m; K pairs contiguous -> two b128 LDS loads
        union { v16bf v; v8bf h[2]; } afrag;
        afrag.h[0] = *(const v8bf*)&xs[wave * 16 + m][kb];
        afrag.h[1] = *(const v8bf*)&xs[wave * 16 + m][16 + kb];

#pragma unroll
        for (int c = 0; c < NCT; ++c) {
            union { v16bf v; v8bf h[2]; } bfrag;
            bfrag.h[0] = *(const v8bf*)&ws[c * 16 + m][kb];
            bfrag.h[1] = *(const v8bf*)&ws[c * 16 + m][16 + kb];
            acc[c] = __builtin_amdgcn_wmma_f32_16x16x32_bf16(
                false, afrag.v, false, bfrag.v, (short)0, acc[c], false, false);
        }
    }

    // Epilogue: rs[r] = sum over this lane's columns of tanh(acc + b[col]) * u[col]
    float rs[8];
#pragma unroll
    for (int r = 0; r < 8; ++r) rs[r] = 0.f;
#pragma unroll
    for (int c = 0; c < NCT; ++c) {
        const int col = c * 16 + m;
        const float bv = bias[col];
        const float uv = u[col];
#pragma unroll
        for (int r = 0; r < 8; ++r)
            rs[r] += tanhf(acc[c][r] + bv) * uv;
    }
    // Reduce across the 16 lanes of each half (rows r in lanes 0-15, r+8 in 16-31)
#pragma unroll
    for (int msk = 1; msk <= 8; msk <<= 1)
#pragma unroll
        for (int r = 0; r < 8; ++r)
            rs[r] += __shfl_xor(rs[r], msk, 32);

    if ((lane & 15) == 0) {
        const int half = lane >> 4;  // 0 -> rows 0..7, 1 -> rows 8..15
#pragma unroll
        for (int r = 0; r < 8; ++r) {
            long row = rowbase + wave * 16 + half * 8 + r;
            ait[row] = expf(rs[r]) * (float)mask[row];
        }
    }
}

// ---------------------------------------------------------------------------
// Kernel 2: inv_denom[b] = 1 / (sum_s ait[b,s] + EPS)
// ---------------------------------------------------------------------------
__global__ __launch_bounds__(256) void denom_kernel(
    const float* __restrict__ ait, float* __restrict__ inv_denom)
{
    __shared__ float red[256];
    const int b = blockIdx.x;
    float s = 0.f;
    for (int i = threadIdx.x; i < S_; i += 256) s += ait[(long)b * S_ + i];
    red[threadIdx.x] = s;
    __syncthreads();
    for (int off = 128; off > 0; off >>= 1) {
        if (threadIdx.x < off) red[threadIdx.x] += red[threadIdx.x + off];
        __syncthreads();
    }
    if (threadIdx.x == 0) inv_denom[b] = 1.0f / (red[0] + 1e-7f);
}

// ---------------------------------------------------------------------------
// Kernel 3: out[b,d] = (sum_s x[b,s,d] * ait[b,s]) * inv_denom[b]   (pure f32)
// ---------------------------------------------------------------------------
__global__ __launch_bounds__(256) void pool_kernel(
    const float* __restrict__ x, const float* __restrict__ ait,
    const float* __restrict__ inv_denom, float* __restrict__ out)
{
    const int b    = blockIdx.x / (D_ / 256);
    const int dblk = blockIdx.x % (D_ / 256);
    const int d    = dblk * 256 + threadIdx.x;
    __shared__ float a_s[256];

    const float* xb = x + (long)b * S_ * D_;
    float acc = 0.f;
    for (int s0 = 0; s0 < S_; s0 += 256) {
        __syncthreads();
        a_s[threadIdx.x] = ait[(long)b * S_ + s0 + threadIdx.x];
        __syncthreads();
#pragma unroll 8
        for (int i = 0; i < 256; ++i)
            acc += xb[(long)(s0 + i) * D_ + d] * a_s[i];
    }
    out[(long)b * D_ + d] = acc * inv_denom[b];
}

// ---------------------------------------------------------------------------
extern "C" void kernel_launch(void* const* d_in, const int* in_sizes, int n_in,
                              void* d_out, int out_size, void* d_ws, size_t ws_size,
                              hipStream_t stream) {
    const float* x    = (const float*)d_in[0];
    const float* W    = (const float*)d_in[1];
    const float* bias = (const float*)d_in[2];
    const float* u    = (const float*)d_in[3];
    const int*   mask = (const int*)d_in[4];

    float* ait       = (float*)d_ws;        // ROWS floats (512 KB)
    float* inv_denom = ait + ROWS;          // B_ floats
    float* out       = (float*)d_out;       // [B, D]

    attn_scores_kernel<<<ROWS / WGROWS, 256, 0, stream>>>(x, W, bias, u, mask, ait);
    denom_kernel<<<B_, 256, 0, stream>>>(ait, inv_denom);
    pool_kernel<<<B_ * (D_ / 256), 256, 0, stream>>>(x, ait, inv_denom, out);
}
